// RPN_10617159156272
// MI455X (gfx1250) — matrix operand order repaired and checked by reading, not compiled, not run on hardware
//
#include <hip/hip_runtime.h>
#include <hip/hip_bf16.h>

// ---------------------------------------------------------------------------
// Types for CDNA5 WMMA
// ---------------------------------------------------------------------------
typedef __attribute__((ext_vector_type(16))) __bf16 v16bf;
typedef __attribute__((ext_vector_type(8)))  float  v8f;

__device__ __forceinline__ float bf2f(unsigned short u) {
    return __uint_as_float(((unsigned int)u) << 16);
}
__device__ __forceinline__ unsigned short f2bf(float f) {
    unsigned int b = __float_as_uint(f);
    return (unsigned short)((b + 0x7fffu + ((b >> 16) & 1u)) >> 16);  // RNE
}

// ---------------------------------------------------------------------------
// Constants (FPN levels)
// ---------------------------------------------------------------------------
#define NLV 5
#define KTOT 4819          // 819 + 4*1000
#define POST_NMS 300
#define NMS_THR 0.7f
#define NEG_SCORE -1.0e9f

// ---------------------------------------------------------------------------
// Kernel 1: conv weights (256,256,3,3) f32 -> bf16, same layout Wt[co][kk].
// ---------------------------------------------------------------------------
__global__ void pack_weights(const float* __restrict__ w, unsigned short* __restrict__ Wt) {
    int i = blockIdx.x * blockDim.x + threadIdx.x;
    if (i >= 2304 * 256) return;
    Wt[i] = f2bf(w[i]);
}

// ---------------------------------------------------------------------------
// Implicit-GEMM conv: im2col staging of a 128(m) x 32(k) activation tile.
// Each thread stages two chunks of 8 consecutive K for one spatial row:
// 8 independent global_load_b32 in flight -> pack -> one ds_store_b128.
// ---------------------------------------------------------------------------
__device__ __forceinline__ void stage_im2col(
    unsigned short* __restrict__ Bs, const float* __restrict__ fin,
    int tid, int blkm, int k0, int H, int W, int HWs, int M)
{
    #pragma unroll
    for (int jj = 0; jj < 2; ++jj) {
        int q   = tid + jj * 256;   // 0..511 chunks
        int ml  = q >> 2;           // 0..127 spatial row in tile
        int klb = (q & 3) << 3;     // 0,8,16,24
        int m   = blkm + ml;
        float v[8];
        if (m < M) {
            int n   = m / HWs;
            int hwp = m - n * HWs;
            int y   = hwp / W;
            int x   = hwp - y * W;
            int kk  = k0 + klb;
            int ci  = kk / 9;
            int r9  = kk - ci * 9;
            const float* fb = fin + (size_t)(n * 256) * HWs;
            #pragma unroll
            for (int e = 0; e < 8; ++e) {
                int ky = r9 / 3;
                int kx = r9 - ky * 3;
                int yy = y + ky - 1;
                int xx = x + kx - 1;
                bool inb = (yy >= 0) & (yy < H) & (xx >= 0) & (xx < W);
                v[e] = inb ? fb[(size_t)ci * HWs + yy * W + xx] : 0.0f;
                if (++r9 == 9) { r9 = 0; ++ci; }
            }
        } else {
            #pragma unroll
            for (int e = 0; e < 8; ++e) v[e] = 0.0f;
        }
        unsigned int pk0 = (unsigned int)f2bf(v[0]) | ((unsigned int)f2bf(v[1]) << 16);
        unsigned int pk1 = (unsigned int)f2bf(v[2]) | ((unsigned int)f2bf(v[3]) << 16);
        unsigned int pk2 = (unsigned int)f2bf(v[4]) | ((unsigned int)f2bf(v[5]) << 16);
        unsigned int pk3 = (unsigned int)f2bf(v[6]) | ((unsigned int)f2bf(v[7]) << 16);
        *(uint4*)&Bs[ml * 32 + klb] = make_uint4(pk0, pk1, pk2, pk3);
    }
}

// ---------------------------------------------------------------------------
// Kernel 2: 3x3 conv 256->256 + bias + ReLU as implicit GEMM via WMMA bf16.
// GEMM-M = out channel (64/block), GEMM-N = spatial (128/block), K = 2304.
// 256 threads = 8 waves; wave (wm,wn) owns 16(co) x 64(m) = 4 fragments.
// A (weights) fragments loaded directly from global (L2-resident, two
// contiguous 16B chunks per lane). B tile double-buffered in LDS.
// All four B fragments are preloaded, then the 4 WMMAs issue back-to-back.
// ---------------------------------------------------------------------------
__global__ __launch_bounds__(256) void conv3x3_wmma(
    const float* __restrict__ fin,          // (2,256,H,W) f32
    const unsigned short* __restrict__ Wt,  // bf16 [co=256][kk=2304]
    const float* __restrict__ bias,         // (256,)
    unsigned short* __restrict__ tout,      // bf16 (2,256,H,W)
    int H, int W, int M)                    // M = 2*H*W
{
    __shared__ __align__(16) unsigned short Bs[2][128 * 32];  // 2 x 8KB

    const int tid  = threadIdx.x;
    const int blkm = blockIdx.x * 128;      // spatial tile base
    const int blkc = blockIdx.y * 64;       // out-channel tile base
    const int lane = tid & 31;
    const int wv   = tid >> 5;
    const int wm   = wv & 3;                // co subtile (16 rows)
    const int wn   = wv >> 2;               // m  subtile (64 cols)
    const int lHi  = lane >> 4;
    const int l16  = lane & 15;
    const int HWs  = H * W;

    v8f c0 = {}, c1 = {}, c2 = {}, c3 = {};

    const unsigned short* wrow = Wt + (size_t)(blkc + wm * 16 + l16) * 2304;

    stage_im2col(Bs[0], fin, tid, blkm, 0, H, W, HWs, M);

    for (int kt = 0; kt < 72; ++kt) {
        __syncthreads();
        const int cur = kt & 1;
        const int k0  = kt * 32;

        // A fragment: two contiguous 16B global loads (contiguous k order, same as B:
        // chunk0 -> K = k0 + lHi*16 + [0..7], chunk1 -> K = k0 + lHi*16 + 8 + [0..7])
        union { v16bf v; uint4 q[2]; } fa;
        const unsigned short* ab = wrow + k0 + lHi * 16;
        fa.q[0] = *(const uint4*)(ab);
        fa.q[1] = *(const uint4*)(ab + 8);

        // Preload ALL B fragments (8x ds_load_b128 issued as a batch)
        union { v16bf v; uint4 q[2]; } fb0, fb1, fb2, fb3;
        {
            const unsigned short* bbase =
                &Bs[cur][((wn * 64 + l16) * 32) + lHi * 16];
            fb0.q[0] = *(const uint4*)(bbase);
            fb0.q[1] = *(const uint4*)(bbase + 8);
            fb1.q[0] = *(const uint4*)(bbase + 16 * 32);
            fb1.q[1] = *(const uint4*)(bbase + 16 * 32 + 8);
            fb2.q[0] = *(const uint4*)(bbase + 32 * 32);
            fb2.q[1] = *(const uint4*)(bbase + 32 * 32 + 8);
            fb3.q[0] = *(const uint4*)(bbase + 48 * 32);
            fb3.q[1] = *(const uint4*)(bbase + 48 * 32 + 8);
        }

        // overlap next-tile staging with the in-flight fragment loads
        if (kt + 1 < 72)
            stage_im2col(Bs[cur ^ 1], fin, tid, blkm, k0 + 32, H, W, HWs, M);

        // 4 WMMAs back-to-back on the XDL pipe, reusing fa
        c0 = __builtin_amdgcn_wmma_f32_16x16x32_bf16(false, fa.v, false, fb0.v,
                                                     (short)0, c0, false, false);
        c1 = __builtin_amdgcn_wmma_f32_16x16x32_bf16(false, fa.v, false, fb1.v,
                                                     (short)0, c1, false, false);
        c2 = __builtin_amdgcn_wmma_f32_16x16x32_bf16(false, fa.v, false, fb2.v,
                                                     (short)0, c2, false, false);
        c3 = __builtin_amdgcn_wmma_f32_16x16x32_bf16(false, fa.v, false, fb3.v,
                                                     (short)0, c3, false, false);
    }

    // ---- epilogue: bias + ReLU + bf16 store (lanes -> consecutive x) ----
    #pragma unroll
    for (int r = 0; r < 8; ++r) {
        int co  = blkc + wm * 16 + r + lHi * 8;
        float bv = bias[co];
        float vals[4] = {c0[r], c1[r], c2[r], c3[r]};
        #pragma unroll
        for (int fidx = 0; fidx < 4; ++fidx) {
            int m = blkm + wn * 64 + fidx * 16 + l16;
            if (m < M) {
                float v = vals[fidx] + bv;
                v = v > 0.0f ? v : 0.0f;
                int n   = m / HWs;
                int hwp = m - n * HWs;
                tout[(size_t)(n * 256 + co) * HWs + hwp] = f2bf(v);
            }
        }
    }
}

// ---------------------------------------------------------------------------
// Kernel 3: 1x1 heads: cls(3) + bbox mean(12) per position; sigmoid scores.
// scores layout: [n][hw*3 + a]; deltas: same index * 4.
// ---------------------------------------------------------------------------
__global__ void rpn_heads(const unsigned short* __restrict__ t,
                          const float* __restrict__ cls_w, const float* __restrict__ cls_b,
                          const float* __restrict__ bbox_w, const float* __restrict__ bbox_b,
                          float* __restrict__ scores, float* __restrict__ deltas,
                          int H, int W)
{
    int HWs = H * W;
    int m = blockIdx.x * blockDim.x + threadIdx.x;
    if (m >= 2 * HWs) return;
    int n   = m / HWs;
    int hwp = m - n * HWs;

    float acc[15];
    #pragma unroll
    for (int o = 0; o < 15; ++o) acc[o] = 0.0f;

    const unsigned short* tp = t + (size_t)(n * 256) * HWs + hwp;
    for (int ci = 0; ci < 256; ++ci) {
        float v = bf2f(tp[(size_t)ci * HWs]);
        #pragma unroll
        for (int a = 0; a < 3; ++a)
            acc[a] += v * cls_w[a * 256 + ci];
        #pragma unroll
        for (int a = 0; a < 3; ++a)
            #pragma unroll
            for (int j = 0; j < 4; ++j)
                acc[3 + a * 4 + j] += v * bbox_w[(a * 8 + j) * 256 + ci];
    }
    int base = n * (HWs * 3) + hwp * 3;
    #pragma unroll
    for (int a = 0; a < 3; ++a) {
        float s = 1.0f / (1.0f + __expf(-(acc[a] + cls_b[a])));
        scores[base + a] = s;
        #pragma unroll
        for (int j = 0; j < 4; ++j)
            deltas[(base + a) * 4 + j] = acc[3 + a * 4 + j] + bbox_b[a * 8 + j];
    }
}

// ---------------------------------------------------------------------------
// Kernel 4: deterministic top-k selection per (batch, level) via histogram
// threshold + stable prefix-sum compaction. One block per batch.
// ---------------------------------------------------------------------------
#define TKB 512
#define NBINS 1024
__global__ __launch_bounds__(TKB) void topk_select(
    const float* __restrict__ scores, int HWA, int k, int* __restrict__ sel)
{
    int b = blockIdx.x;
    const float* s = scores + (size_t)b * HWA;
    __shared__ int hist[NBINS];
    __shared__ int cA[TKB], cE[TKB];
    __shared__ int s_thr, s_above;
    int tid = threadIdx.x;

    for (int i = tid; i < NBINS; i += TKB) hist[i] = 0;
    __syncthreads();
    for (int i = tid; i < HWA; i += TKB) {
        int bin = (int)(s[i] * NBINS);
        bin = bin < 0 ? 0 : (bin > NBINS - 1 ? NBINS - 1 : bin);
        atomicAdd(&hist[bin], 1);
    }
    __syncthreads();
    if (tid == 0) {
        int cum = 0, bin = NBINS - 1;
        for (; bin > 0; --bin) { cum += hist[bin]; if (cum >= k) break; }
        if (cum < k) cum += hist[0], bin = 0;
        s_thr = bin;
        s_above = cum - hist[bin];
    }
    __syncthreads();
    int thr = s_thr, above = s_above;

    int myA = 0, myE = 0;
    for (int i = tid; i < HWA; i += TKB) {
        int bin = (int)(s[i] * NBINS);
        bin = bin < 0 ? 0 : (bin > NBINS - 1 ? NBINS - 1 : bin);
        if (bin > thr) ++myA; else if (bin == thr) ++myE;
    }
    cA[tid] = myA; cE[tid] = myE;
    __syncthreads();
    if (tid == 0) {  // exclusive scans (serial, deterministic)
        int sa = 0, se = 0;
        for (int i = 0; i < TKB; ++i) {
            int a = cA[i]; cA[i] = sa; sa += a;
            int e = cE[i]; cE[i] = se; se += e;
        }
    }
    __syncthreads();
    int pa = cA[tid], pe = cE[tid];
    for (int i = tid; i < HWA; i += TKB) {
        int bin = (int)(s[i] * NBINS);
        bin = bin < 0 ? 0 : (bin > NBINS - 1 ? NBINS - 1 : bin);
        if (bin > thr) {
            sel[b * k + pa] = i; ++pa;
        } else if (bin == thr) {
            if (pe < k - above) sel[b * k + above + pe] = i;
            ++pe;
        }
    }
}

// ---------------------------------------------------------------------------
// Kernel 5: decode anchors + deltas, clip, fill concat boxes/scores.
// ---------------------------------------------------------------------------
__global__ void decode_boxes(const int* __restrict__ sel,
                             const float* __restrict__ scores,
                             const float* __restrict__ deltas,
                             const float* __restrict__ im_info,
                             float* __restrict__ boxes_all,
                             float* __restrict__ scores_all)
{
    const int kW[NLV]   = {21, 42, 84, 168, 336};
    const int kK[NLV]   = {819, 1000, 1000, 1000, 1000};
    const int kG[NLV]   = {0, 819, 1819, 2819, 3819};
    const int kSC[NLV]  = {0, 1638, 7938, 33138, 133938};   // 6*cumHW
    const int kSEL[NLV] = {0, 1638, 3638, 5638, 7638};      // 2*cumK

    int g = blockIdx.x * blockDim.x + threadIdx.x;
    if (g >= 2 * KTOT) return;
    int b = g / KTOT;
    int slot = g - b * KTOT;
    int lvl = NLV - 1;
    #pragma unroll
    for (int i = 0; i < NLV; ++i)
        if (slot < kG[i] + kK[i]) { lvl = i; break; }
    int sl  = slot - kG[lvl];
    int W   = kW[lvl];
    int HWA = 0;
    { const int kHW[NLV] = {273, 1050, 4200, 16800, 67200}; HWA = kHW[lvl] * 3; }
    int k   = kK[lvl];

    int idx = sel[kSEL[lvl] + b * k + sl];
    float sc = scores[kSC[lvl] + b * HWA + idx];
    const float* d = deltas + (size_t)(kSC[lvl] + b * HWA + idx) * 4;

    int a   = idx % 3;
    int hwp = idx / 3;
    int y   = hwp / W;
    int x   = hwp - y * W;
    float off    = (float)(1 << (NLV - 1 - lvl));
    float stride = 4.0f * off;
    float size   = 16.0f * off;
    float ratio  = (float)(a + 1);
    float wa = size * sqrtf(1.0f / ratio);
    float ha = size * sqrtf(ratio);
    float cxa = ((float)x + 0.5f) * stride;
    float cya = ((float)y + 0.5f) * stride;
    float cx = d[0] * wa + cxa;
    float cy = d[1] * ha + cya;
    float w2 = __expf(d[2]) * wa;
    float h2 = __expf(d[3]) * ha;
    float imh = im_info[b * 6 + 0] - 1.0f;
    float imw = im_info[b * 6 + 1] - 1.0f;
    float x1 = fminf(fmaxf(cx - 0.5f * w2, 0.0f), imw);
    float y1 = fminf(fmaxf(cy - 0.5f * h2, 0.0f), imh);
    float x2 = fminf(fmaxf(cx + 0.5f * w2, 0.0f), imw);
    float y2 = fminf(fmaxf(cy + 0.5f * h2, 0.0f), imh);

    float* bo = boxes_all + (size_t)(b * KTOT + slot) * 4;
    bo[0] = x1; bo[1] = y1; bo[2] = x2; bo[3] = y2;
    scores_all[b * KTOT + slot] = sc;
}

// ---------------------------------------------------------------------------
// Kernel 6: greedy NMS, 300 iterations, one block per image.
// Boxes + areas cached in registers; scores in LDS.
// ---------------------------------------------------------------------------
#define NMSB 512
#define PERT 10   // ceil(4819/512)
__global__ __launch_bounds__(NMSB) void nms_kernel(
    const float* __restrict__ boxes_all, const float* __restrict__ scores_all,
    int* __restrict__ keep)
{
    int b = blockIdx.x;
    __shared__ float s[KTOT];
    __shared__ float redV[NMSB];
    __shared__ int   redI[NMSB];
    __shared__ float bj[4];
    __shared__ float areaj;
    int tid = threadIdx.x;

    float bx0[PERT], bx1[PERT], bx2[PERT], bx3[PERT], ar[PERT];
    #pragma unroll
    for (int j = 0; j < PERT; ++j) {
        int i = tid + j * NMSB;
        if (i < KTOT) {
            const float* bp = boxes_all + (size_t)(b * KTOT + i) * 4;
            bx0[j] = bp[0]; bx1[j] = bp[1]; bx2[j] = bp[2]; bx3[j] = bp[3];
            ar[j]  = (bx2[j] - bx0[j]) * (bx3[j] - bx1[j]);
            s[i]   = scores_all[b * KTOT + i];
        }
    }
    __syncthreads();

    for (int it = 0; it < POST_NMS; ++it) {
        float bv = -1.0e30f; int bi = KTOT;
        #pragma unroll
        for (int j = 0; j < PERT; ++j) {
            int i = tid + j * NMSB;
            if (i < KTOT) {
                float v = s[i];
                if (v > bv || (v == bv && i < bi)) { bv = v; bi = i; }
            }
        }
        redV[tid] = bv; redI[tid] = bi;
        __syncthreads();
        for (int st = NMSB / 2; st > 0; st >>= 1) {
            if (tid < st) {
                float v2 = redV[tid + st]; int i2 = redI[tid + st];
                if (v2 > redV[tid] || (v2 == redV[tid] && i2 < redI[tid])) {
                    redV[tid] = v2; redI[tid] = i2;
                }
            }
            __syncthreads();
        }
        int jsel = redI[0];
        if (tid == 0) {
            keep[b * POST_NMS + it] = jsel;
            const float* bp = boxes_all + (size_t)(b * KTOT + jsel) * 4;
            bj[0] = bp[0]; bj[1] = bp[1]; bj[2] = bp[2]; bj[3] = bp[3];
            areaj = (bj[2] - bj[0]) * (bj[3] - bj[1]);
        }
        __syncthreads();
        float d0 = bj[0], d1 = bj[1], d2 = bj[2], d3 = bj[3], aj = areaj;
        #pragma unroll
        for (int j = 0; j < PERT; ++j) {
            int i = tid + j * NMSB;
            if (i < KTOT) {
                float xx1 = fmaxf(bx0[j], d0), yy1 = fmaxf(bx1[j], d1);
                float xx2 = fminf(bx2[j], d2), yy2 = fminf(bx3[j], d3);
                float inter = fmaxf(xx2 - xx1, 0.0f) * fmaxf(yy2 - yy1, 0.0f);
                float iou = inter / (ar[j] + aj - inter + 1e-6f);
                if (iou > NMS_THR) s[i] = NEG_SCORE;
            }
        }
        __syncthreads();
    }
}

// ---------------------------------------------------------------------------
// Kernel 7: final gather -> (600, 5): [batch_idx, x1,y1,x2,y2]
// ---------------------------------------------------------------------------
__global__ void write_out(const float* __restrict__ boxes_all,
                          const int* __restrict__ keep, float* __restrict__ out)
{
    int i = blockIdx.x * blockDim.x + threadIdx.x;
    if (i >= 2 * POST_NMS) return;
    int b = i / POST_NMS;
    int j = keep[i];
    out[i * 5 + 0] = (float)b;
    const float* bp = boxes_all + (size_t)(b * KTOT + j) * 4;
    out[i * 5 + 1] = bp[0];
    out[i * 5 + 2] = bp[1];
    out[i * 5 + 3] = bp[2];
    out[i * 5 + 4] = bp[3];
}

// ---------------------------------------------------------------------------
// Host launcher
// ---------------------------------------------------------------------------
extern "C" void kernel_launch(void* const* d_in, const int* in_sizes, int n_in,
                              void* d_out, int out_size, void* d_ws, size_t ws_size,
                              hipStream_t stream) {
    (void)in_sizes; (void)n_in; (void)out_size; (void)ws_size;

    static const int LH[NLV]  = {13, 25, 50, 100, 200};
    static const int LW[NLV]  = {21, 42, 84, 168, 336};
    static const int LHW[NLV] = {273, 1050, 4200, 16800, 67200};
    static const int CUMHW[NLV] = {0, 273, 1323, 5523, 22323};
    static const int LK[NLV]  = {819, 1000, 1000, 1000, 1000};
    static const int SELOFF[NLV] = {0, 1638, 3638, 5638, 7638};
    const int TOTHW = 89523;

    const float* f[NLV];
    for (int i = 0; i < NLV; ++i) f[i] = (const float*)d_in[i];
    const float* im_info = (const float*)d_in[5];
    const float* conv_w  = (const float*)d_in[6];
    const float* conv_b  = (const float*)d_in[7];
    const float* cls_w   = (const float*)d_in[8];
    const float* cls_b   = (const float*)d_in[9];
    const float* bbox_w  = (const float*)d_in[10];
    const float* bbox_b  = (const float*)d_in[11];
    float* out = (float*)d_out;

    // Workspace carve-out
    char* p = (char*)d_ws;
    auto carve = [&](size_t bytes) { char* r = p; p += (bytes + 255) & ~(size_t)255; return (void*)r; };
    unsigned short* Wt    = (unsigned short*)carve((size_t)2304 * 256 * 2);
    unsigned short* tbuf  = (unsigned short*)carve((size_t)2 * 256 * TOTHW * 2);
    float* scores         = (float*)carve((size_t)2 * 3 * TOTHW * 4);
    float* deltas         = (float*)carve((size_t)2 * 3 * TOTHW * 16);
    int*   sel            = (int*)carve((size_t)2 * KTOT * 4);
    float* boxes_all      = (float*)carve((size_t)2 * KTOT * 4 * 4);
    float* scores_all     = (float*)carve((size_t)2 * KTOT * 4);
    int*   keep           = (int*)carve((size_t)2 * POST_NMS * 4);

    // 1. pack weights to bf16 (identity layout [co][kk])
    pack_weights<<<(2304 * 256 + 255) / 256, 256, 0, stream>>>(conv_w, Wt);

    // 2. conv3x3 + ReLU per level (WMMA)
    for (int i = 0; i < NLV; ++i) {
        int M = 2 * LHW[i];
        dim3 grid((M + 127) / 128, 4);
        conv3x3_wmma<<<grid, 256, 0, stream>>>(
            f[i], Wt, conv_b, tbuf + (size_t)2 * 256 * CUMHW[i], LH[i], LW[i], M);
    }

    // 3. heads per level
    for (int i = 0; i < NLV; ++i) {
        int M = 2 * LHW[i];
        rpn_heads<<<(M + 255) / 256, 256, 0, stream>>>(
            tbuf + (size_t)2 * 256 * CUMHW[i], cls_w, cls_b, bbox_w, bbox_b,
            scores + (size_t)6 * CUMHW[i], deltas + (size_t)24 * CUMHW[i],
            LH[i], LW[i]);
    }

    // 4. deterministic top-k per (batch, level)
    for (int i = 0; i < NLV; ++i) {
        topk_select<<<2, TKB, 0, stream>>>(
            scores + (size_t)6 * CUMHW[i], LHW[i] * 3, LK[i], sel + SELOFF[i]);
    }

    // 5. decode + clip + concat
    decode_boxes<<<(2 * KTOT + 255) / 256, 256, 0, stream>>>(
        sel, scores, deltas, im_info, boxes_all, scores_all);

    // 6. NMS (one block per image)
    nms_kernel<<<2, NMSB, 0, stream>>>(boxes_all, scores_all, keep);

    // 7. output gather
    write_out<<<(2 * POST_NMS + 255) / 256, 256, 0, stream>>>(boxes_all, keep, out);
}
